// BioRNN_58179626992359
// MI455X (gfx1250) — matrix operand order repaired and checked by reading, not compiled, not run on hardware
//
#include <hip/hip_runtime.h>

// ---------------------------------------------------------------------------
// BioRNN scan for MI455X (gfx1250, wave32, WMMA).
// 4 workgroups x 256 threads; each WG owns a 16-row batch tile for all T steps.
// Recurrent + input GEMM fused into one 18-K-tile bf16 WMMA loop per step.
// Weights stream from L2 each step (laundered offset blocks LICM/spill while
// keeping the global address space -> global_load_b128, LOADcnt only).
// Noise double-buffered into LDS one step ahead via async-to-LDS (ASYNCcnt).
// ---------------------------------------------------------------------------

typedef __attribute__((ext_vector_type(16))) __bf16 v16bf;
typedef __attribute__((ext_vector_type(8)))  float  v8f;

constexpr int   Bb = 64, T = 2048, I = 64, H = 512, O = 64;
constexpr float TAU = 0.1f;
constexpr int   NT     = H / 16;          // 32 N-tiles over hidden dim
constexpr int   KT_REC = H / 32;          // 16 K-tiles for W_rec
constexpr int   KT_ALL = KT_REC + I / 32; // 18 K-tiles: [W_rec | W_in]
constexpr int   AW     = H + I;           // 576: A-buffer width [tanh(x) | inp_t]

#if defined(__has_builtin)
#if __has_builtin(__builtin_amdgcn_global_load_async_to_lds_b32) && \
    __has_builtin(__builtin_amdgcn_s_wait_asynccnt)
#define HAVE_ASYNC_LDS 1
#endif
#endif
#ifndef HAVE_ASYNC_LDS
#define HAVE_ASYNC_LDS 0
#endif

typedef __attribute__((address_space(1))) int gint_t;
typedef __attribute__((address_space(3))) int lint_t;

union FragBF {
  uint4 u[2];
  v16bf v;
};

__device__ __forceinline__ unsigned short f2bf(float f) {
  unsigned int u = __builtin_bit_cast(unsigned int, f);
  u += 0x7FFFu + ((u >> 16) & 1u);   // round-to-nearest-even
  return (unsigned short)(u >> 16);
}

__device__ __forceinline__ float fast_tanh(float x) {
  // tanh(x) = 1 - 2/(exp2(2x*log2 e)+1); saturates correctly at +-inf.
  float e = __builtin_amdgcn_exp2f(x * 2.8853900817779268f);
  return 1.0f - 2.0f * __builtin_amdgcn_rcpf(e + 1.0f);
}

// ---------------------------------------------------------------------------
// Pack [W_rec | W_in] as bf16 WMMA B-fragments.
// Fragment f = nt*KT_ALL + kt, 1KB each: lane l -> column N = l&15,
// element e -> K = e + 16*(l>>4). Flat ushort index = f*512 + l*16 + e.
// ---------------------------------------------------------------------------
__global__ __launch_bounds__(256) void pack_weights(
    const float* __restrict__ W_in, const float* __restrict__ W_rec,
    unsigned short* __restrict__ wpack) {
  int gid = blockIdx.x * 256 + threadIdx.x;      // 0 .. 32*18*512-1
  int f = gid >> 9, r = gid & 511;
  int l = r >> 4, e = r & 15;
  int nt = f / KT_ALL, kt = f % KT_ALL;
  int N = l & 15, K = e + 16 * (l >> 4);
  float v;
  if (kt < KT_REC) v = W_rec[(nt * 16 + N) * H + kt * 32 + K];
  else             v = W_in[(nt * 16 + N) * I + (kt - KT_REC) * 32 + K];
  wpack[gid] = f2bf(v);
}

__global__ __launch_bounds__(256) void pack_wout(
    const float* __restrict__ W_out, unsigned short* __restrict__ wopack) {
  int gid = blockIdx.x * 256 + threadIdx.x;      // 0 .. 4*16*512-1
  int f = gid >> 9, r = gid & 511;
  int l = r >> 4, e = r & 15;
  int nt = f >> 4, kt = f & 15;
  int N = l & 15, K = e + 16 * (l >> 4);
  wopack[gid] = f2bf(W_out[(nt * 16 + N) * H + kt * 32 + K]);
}

__global__ __launch_bounds__(256) void copy_l2(
    const float* __restrict__ W_in, const float* __restrict__ W_out,
    float* __restrict__ dst) {
  int gid = blockIdx.x * 256 + threadIdx.x;      // 0 .. 65535
  dst[gid] = (gid < H * I) ? W_in[gid] : W_out[gid - H * I];
}

// ---------------------------------------------------------------------------
// Main persistent scan kernel: grid = 4 WGs (one 16-row batch tile each).
// ---------------------------------------------------------------------------
__global__ __launch_bounds__(256) void biornn_main(
    const float* __restrict__ input,   // [B,T,I]
    const float* __restrict__ noise,   // [B,T,H]
    const float* __restrict__ bias,    // [1,H]
    const unsigned short* __restrict__ wpack,
    const unsigned short* __restrict__ wopack,
    float* __restrict__ outp,          // [B,T,O]
    float* __restrict__ hid) {         // [B,T,H]
  __shared__ float          xs[16 * H];    // f32 state x, 32 KB
  __shared__ unsigned short abf[16 * AW];  // bf16 [tanh(x) | inp_t], 18 KB
  __shared__ unsigned short xbf[16 * H];   // bf16 x_new for out-proj, 16 KB
#if HAVE_ASYNC_LDS
  __shared__ float nbuf[2][256 * 32];      // noise double buffer, 64 KB
#endif

  const int td  = threadIdx.x;
  const int w   = td >> 5;     // wave id 0..7
  const int l   = td & 31;     // lane
  const int lh  = l >> 4;      // lane-half select
  const int ln  = l & 15;
  const int g16 = blockIdx.x * 16;

  // per-lane bias for this wave's 4 N-tiles (constant over t)
  float bcol[4];
  for (int q = 0; q < 4; ++q) bcol[q] = bias[(w * 4 + q) * 16 + ln];

  // Issue one step of noise prefetch: each thread fetches exactly the 32
  // values it will consume in phase C of step tt (thread-private LDS region,
  // so no barrier is needed -- only s_wait_asynccnt).
#if HAVE_ASYNC_LDS
  auto prefetch_noise = [&](int tt) {
    if (tt >= T) return;
    const int slot = tt & 1;
    for (int q = 0; q < 4; ++q) {
      const int col = (w * 4 + q) * 16 + ln;
      for (int r = 0; r < 8; ++r) {
        const int m = r + 8 * lh;
        const size_t gidx = ((size_t)(g16 + m) * T + tt) * H + col;
        __builtin_amdgcn_global_load_async_to_lds_b32(
            (gint_t*)(uintptr_t)(noise + gidx),
            (lint_t*)(unsigned)(uintptr_t)&nbuf[slot][td * 32 + q * 8 + r],
            0, 0);
      }
    }
  };
  prefetch_noise(0);
#endif

  // x0 = 0
  {
    const int m = td >> 4, c0 = (td & 15) * 32;
    for (int j = 0; j < 32; ++j) xs[m * H + c0 + j] = 0.0f;
  }
  __syncthreads();

  for (int t = 0; t < T; ++t) {
    // Launder a zero offset (not the pointer!) every step: the weight-stream
    // addresses become loop-variant so LICM cannot hoist 576 fragments into
    // VGPRs/scratch (round-1 pathology), yet the base pointers keep their
    // promoted global address space so loads lower to global_load_b128
    // (LOADcnt only) instead of flat_load_b128 (LOADcnt+DScnt, LDS-path
    // contention with the A-fragment ds_loads).
    unsigned toff = 0;
    asm volatile("" : "+s"(toff));

#if HAVE_ASYNC_LDS
    prefetch_noise(t + 1);   // overlap next step's HBM pull with this GEMM
#endif

    // ---- phase A: abf = [ bf16(tanh(x)) | bf16(input_t) ] ----
    {
      const int m = td >> 4, c0 = (td & 15) * 32;
      for (int j = 0; j < 32; ++j)
        abf[m * AW + c0 + j] = f2bf(fast_tanh(xs[m * H + c0 + j]));
      const int i0 = (td & 15) * 4;
      const float4 v =
          *(const float4*)&input[((size_t)(g16 + m) * T + t) * I + i0];
      abf[m * AW + H + i0 + 0] = f2bf(v.x);
      abf[m * AW + H + i0 + 1] = f2bf(v.y);
      abf[m * AW + H + i0 + 2] = f2bf(v.z);
      abf[m * AW + H + i0 + 3] = f2bf(v.w);
    }
    __syncthreads();

    // ---- phase B: acc = tanh(x)@W_rec^T + inp_t@W_in^T (bf16 WMMA) ----
    v8f acc[4];
    for (int q = 0; q < 4; ++q) {
      const int nt = w * 4 + q;
      v8f c = {};
      const uint4* bp =
          (const uint4*)(wpack + toff + (size_t)nt * KT_ALL * 512) + l * 2;
      for (int kt = 0; kt < KT_ALL; ++kt) {
        FragBF a, b;
        const unsigned short* ap = &abf[ln * AW + kt * 32 + 16 * lh];
        a.u[0] = *(const uint4*)(ap);       // K 0..7  (+16*lh)
        a.u[1] = *(const uint4*)(ap + 8);   // K 8..15 (+16*lh)
        b.u[0] = bp[0];
        b.u[1] = bp[1];
        bp += 64;  // next 1KB fragment
        c = __builtin_amdgcn_wmma_f32_16x16x32_bf16(
            false, a.v, false, b.v, (short)0, c, false, false);
      }
      acc[q] = c;
    }

    // ---- phase C: x_new = x + TAU*(-x + acc + b + noise); emit state ----
#if HAVE_ASYNC_LDS
    // Oldest 32 async loads (step t) must be done; step t+1's may remain.
    if (t + 1 < T) __builtin_amdgcn_s_wait_asynccnt(32);
    else           __builtin_amdgcn_s_wait_asynccnt(0);
#endif
    for (int q = 0; q < 4; ++q) {
      const int col = (w * 4 + q) * 16 + ln;
      for (int r = 0; r < 8; ++r) {
        const int m = r + 8 * lh;  // C/D layout: VGPR r, lane-half -> row
        const size_t gidx = ((size_t)(g16 + m) * T + t) * H + col;
#if HAVE_ASYNC_LDS
        const float nz = nbuf[t & 1][td * 32 + q * 8 + r];
#else
        const float nz = noise[gidx];
#endif
        const float xo = xs[m * H + col];
        const float xn = xo + TAU * (acc[q][r] + bcol[q] + nz - xo);
        xs[m * H + col]  = xn;
        xbf[m * H + col] = f2bf(xn);
        hid[gidx] = xn;
      }
    }
    __syncthreads();

    // ---- phase D: out = x_new @ W_out^T (waves 0..3, one N-tile each) ----
    if (w < 4) {
      v8f c = {};
      const uint4* bp =
          (const uint4*)(wopack + toff + (size_t)w * 16 * 512) + l * 2;
      for (int kt = 0; kt < 16; ++kt) {
        FragBF a, b;
        const unsigned short* ap = &xbf[ln * H + kt * 32 + 16 * lh];
        a.u[0] = *(const uint4*)(ap);
        a.u[1] = *(const uint4*)(ap + 8);
        b.u[0] = bp[0];
        b.u[1] = bp[1];
        bp += 64;
        c = __builtin_amdgcn_wmma_f32_16x16x32_bf16(
            false, a.v, false, b.v, (short)0, c, false, false);
      }
      for (int r = 0; r < 8; ++r) {
        const int m = r + 8 * lh;
        outp[((size_t)(g16 + m) * T + t) * O + w * 16 + ln] = c[r];
      }
    }
    __syncthreads();
  }
}

// ---------------------------------------------------------------------------
extern "C" void kernel_launch(void* const* d_in, const int* in_sizes, int n_in,
                              void* d_out, int out_size, void* d_ws,
                              size_t ws_size, hipStream_t stream) {
  const float* input = (const float*)d_in[0];
  const float* noise = (const float*)d_in[1];
  const float* W_in  = (const float*)d_in[2];
  const float* W_rec = (const float*)d_in[3];
  const float* W_out = (const float*)d_in[4];
  const float* bias  = (const float*)d_in[5];

  float* outp = (float*)d_out;                 // [B,T,O]
  float* hid  = outp + (size_t)Bb * T * O;     // [B,T,H]
  float* l2   = hid + (size_t)Bb * T * H;      // [H*I + O*H]

  unsigned short* wpack  = (unsigned short*)d_ws;             // 576 KB
  unsigned short* wopack = wpack + (size_t)NT * KT_ALL * 512; // +64 KB

  pack_weights<<<(NT * KT_ALL * 512) / 256, 256, 0, stream>>>(W_in, W_rec,
                                                              wpack);
  pack_wout<<<(4 * 16 * 512) / 256, 256, 0, stream>>>(W_out, wopack);
  copy_l2<<<(H * I + O * H) / 256, 256, 0, stream>>>(W_in, W_out, l2);
  biornn_main<<<4, 256, 0, stream>>>(input, noise, bias, wpack, wopack, outp,
                                     hid);
}
